// WindowAttention_36352603193767
// MI455X (gfx1250) — compile-verified, matrix operand order checked by reading, not run on hardware
//
#include <hip/hip_runtime.h>
#include <hip/hip_bf16.h>

#define DIMC 192
#define NH   6
#define HD   32
#define NTOK 49
#define NP   64
#define NWIN 4096
#define XLD  200                 // 64 x 192 f16 X / O staging, padded stride (400B, 16B mult)
#define QLD  40                  // 64 x 32  f16 per-head Q/K, padded stride (80B)
#define VTLD 72                  // 32 x 64  f16 per-head V^T, padded stride (144B)
#define PLD  72                  // 64 x 64  f16 probabilities (overlay on Q/K)
#define HCH  (2 * NP * QLD + HD * VTLD)   // per-head LDS chunk: 7424 halfs
#define ATT_SCALE 0.17677669529663687f

typedef __attribute__((ext_vector_type(16))) _Float16 v16h;
typedef __attribute__((ext_vector_type(8)))  _Float16 v8h;
typedef __attribute__((ext_vector_type(4)))  _Float16 v4h;
typedef __attribute__((ext_vector_type(8)))  float    v8f;
typedef __attribute__((ext_vector_type(4)))  float    v4f;

__device__ __forceinline__ int kmap(int i, int hi) {
  return ((i < 8) ? i : (i + 8)) + hi * 8;
}

// A-fragment (16x32 f16) / K^T / V^T B-fragment: lane = row (lane&15), halves are
// two contiguous 8-half runs -> two ds_load_b128 + shuffle.
__device__ __forceinline__ v16h ld_frag_rows(const _Float16* p, int ld, int lane) {
  const _Float16* q = p + (lane & 15) * ld + (lane >> 4) * 8;
  v8h lo = *(const v8h*)q;
  v8h hp = *(const v8h*)(q + 16);
  return __builtin_shufflevector(lo, hp, 0, 1, 2, 3, 4, 5, 6, 7, 8, 9, 10, 11,
                                 12, 13, 14, 15);
}

// B-fragment from fragment-packed weights: 32 contiguous bytes per lane,
// fully coalesced across the wave (two global_load_b128).
__device__ __forceinline__ v16h ld_frag_packed(const _Float16* base, int tile,
                                               int lane) {
  const _Float16* q = base + (((size_t)tile * 32 + lane) << 4);
  v8h lo = *(const v8h*)q;
  v8h hp = *(const v8h*)(q + 8);
  return __builtin_shufflevector(lo, hp, 0, 1, 2, 3, 4, 5, 6, 7, 8, 9, 10, 11,
                                 12, 13, 14, 15);
}

__device__ __forceinline__ v8f wmma16(v16h a, v16h b, v8f c) {
  return __builtin_amdgcn_wmma_f32_16x16x32_f16(false, a, false, b, (short)0, c,
                                                false, false);
}

// -------- prep: swizzle f32 weights into f16 fragment-packed tiles; gather the
// relative-position bias into a dense (H,64,64) fp32 tensor with -1e30 masking.
__global__ void winattn_prep(const float* __restrict__ wqkv,
                             const float* __restrict__ wproj,
                             const float* __restrict__ btab,
                             const int* __restrict__ ridx,
                             _Float16* __restrict__ wqkv_p,
                             _Float16* __restrict__ wproj_p,
                             float* __restrict__ biasf) {
  int i = blockIdx.x * blockDim.x + threadIdx.x;
  // packed linear index: i = tile*512 + lane*16 + elem
  int elem = i & 15, lane = (i >> 4) & 31, t = i >> 9;
  int hi = lane >> 4, n = lane & 15;
  int krow = (elem < 8 ? elem : elem + 8) + hi * 8;

  if (i < DIMC * 3 * DIMC) {            // 216 tiles (36 ntiles x 6 ktiles)
    int kt = t % 6, ntile = t / 6;
    wqkv_p[i] = (_Float16)wqkv[(kt * 32 + krow) * (3 * DIMC) + ntile * 16 + n];
  }
  if (i < DIMC * DIMC) {                // 72 tiles (12 ntiles x 6 ktiles)
    int kt = t % 6, ntile = t / 6;
    wproj_p[i] = (_Float16)wproj[(kt * 32 + krow) * DIMC + ntile * 16 + n];
  }
  if (i < NH * NP * NP) {
    int h = i / (NP * NP);
    int rem = i - h * NP * NP;
    int r = rem / NP, c = rem - r * NP;
    float v;
    if (c >= NTOK)      v = -1e30f;     // mask padded key columns
    else if (r >= NTOK) v = 0.0f;       // padded query rows: unused
    else                v = btab[ridx[r * NTOK + c] * NH + h];
    biasf[i] = v;
  }
}

// -------- fused window attention: one workgroup per window, one wave/head ----
__global__ __launch_bounds__(192, 1) void winattn_kernel(
    const float* __restrict__ x, const _Float16* __restrict__ wqkv_p,
    const float* __restrict__ bqkv, const _Float16* __restrict__ wproj_p,
    const float* __restrict__ bproj, const float* __restrict__ biasf,
    float* __restrict__ out) {
  extern __shared__ _Float16 smem[];
  _Float16* XS = smem;                       // 64 x XLD  (x, f16)
  _Float16* HS = XS + NP * XLD;              // per head: Q, K, V^T
  _Float16* OS = HS + NH * HCH;              // 64 x XLD  (attn out, f16)

  const int tid  = threadIdx.x;
  const int wave = tid >> 5;                 // head id / proj column slice
  const int lane = tid & 31;
  const int n    = lane & 15;
  const int hi   = lane >> 4;
  const int blk  = blockIdx.x;

  // Stage 0: x (49x192 f32) -> XS (64x192 f16, zero-padded rows), vectorized
  const float* xb = x + (size_t)blk * NTOK * DIMC;
  for (int idx = tid; idx < NP * 48; idx += 192) {
    int r = idx / 48, c4 = idx - r * 48;
    v4f v = {};
    if (r < NTOK) v = *(const v4f*)(xb + r * DIMC + c4 * 4);
    v4h hv;
#pragma unroll
    for (int j = 0; j < 4; ++j) hv[j] = (_Float16)v[j];
    *(v4h*)(XS + r * XLD + c4 * 4) = hv;
  }
  __syncthreads();

  // Stage 1: per-head QKV projection (Q pre-scaled); V stored transposed
  _Float16* Qs = HS + wave * HCH;
  _Float16* Ks = Qs + NP * QLD;
  _Float16* Vt = Ks + NP * QLD;
#pragma unroll
  for (int mat = 0; mat < 3; ++mat) {
    const int ntb = mat * 12 + wave * 2;     // global 16-col tile base in 576
    __builtin_prefetch(wqkv_p + (size_t)(ntb * 6) * 512 + lane * 64, 0, 1);
    v16h B[2][6];
#pragma unroll
    for (int nt = 0; nt < 2; ++nt)
#pragma unroll
      for (int kt = 0; kt < 6; ++kt)
        B[nt][kt] = ld_frag_packed(wqkv_p, (ntb + nt) * 6 + kt, lane);
    const float* bv = bqkv + mat * DIMC + wave * HD;
    const float b0 = bv[n], b1 = bv[16 + n];
    const float mult = (mat == 0) ? ATT_SCALE : 1.0f;
    for (int mt = 0; mt < 4; ++mt) {
      v16h a[6];
#pragma unroll
      for (int kt = 0; kt < 6; ++kt)
        a[kt] = ld_frag_rows(XS + mt * 16 * XLD + kt * 32, XLD, lane);
      v8f acc0 = {}, acc1 = {};
#pragma unroll
      for (int kt = 0; kt < 6; ++kt) {
        acc0 = wmma16(a[kt], B[0][kt], acc0);
        acc1 = wmma16(a[kt], B[1][kt], acc1);
      }
      if (mat < 2) {
        _Float16* dst = (mat == 0) ? Qs : Ks;
#pragma unroll
        for (int r = 0; r < 8; ++r) {
          int row = mt * 16 + r + hi * 8;
          dst[row * QLD + n]      = (_Float16)((acc0[r] + b0) * mult);
          dst[row * QLD + 16 + n] = (_Float16)((acc1[r] + b1) * mult);
        }
      } else {  // V^T: 8 consecutive tokens per lane -> single b128 store
        v8h p0, p1;
#pragma unroll
        for (int r = 0; r < 8; ++r) {
          p0[r] = (_Float16)(acc0[r] + b0);
          p1[r] = (_Float16)(acc1[r] + b1);
        }
        *(v8h*)(Vt + n * VTLD + mt * 16 + hi * 8)        = p0;
        *(v8h*)(Vt + (16 + n) * VTLD + mt * 16 + hi * 8) = p1;
      }
    }
  }

  // Stage 2: S = Q K^T + bias, row softmax
  v16h kf[4];
#pragma unroll
  for (int nt = 0; nt < 4; ++nt)
    kf[nt] = ld_frag_rows(Ks + nt * 16 * QLD, QLD, lane);  // K^T tiles
  v8f S[4][4];
#pragma unroll
  for (int mt = 0; mt < 4; ++mt) {
    v16h a = ld_frag_rows(Qs + mt * 16 * QLD, QLD, lane);
#pragma unroll
    for (int nt = 0; nt < 4; ++nt) {
      v8f z = {};
      S[mt][nt] = wmma16(a, kf[nt], z);
    }
  }
  const float* bh = biasf + wave * NP * NP;
#pragma unroll
  for (int mt = 0; mt < 4; ++mt) {
#pragma unroll
    for (int r = 0; r < 8; ++r) {
      const int row = mt * 16 + r + hi * 8;
      float v0 = S[mt][0][r] + bh[row * NP + 0 + n];
      float v1 = S[mt][1][r] + bh[row * NP + 16 + n];
      float v2 = S[mt][2][r] + bh[row * NP + 32 + n];
      float v3 = S[mt][3][r] + bh[row * NP + 48 + n];
      float mx = fmaxf(fmaxf(v0, v1), fmaxf(v2, v3));
#pragma unroll
      for (int m = 1; m < 16; m <<= 1) mx = fmaxf(mx, __shfl_xor(mx, m, 32));
      v0 = __expf(v0 - mx); v1 = __expf(v1 - mx);
      v2 = __expf(v2 - mx); v3 = __expf(v3 - mx);
      float sum = v0 + v1 + v2 + v3;
#pragma unroll
      for (int m = 1; m < 16; m <<= 1) sum += __shfl_xor(sum, m, 32);
      const float inv = 1.0f / sum;
      S[mt][0][r] = v0 * inv; S[mt][1][r] = v1 * inv;
      S[mt][2][r] = v2 * inv; S[mt][3][r] = v3 * inv;
    }
  }
  // spill P (f16), overlaying the dead Q/K region (wave-private, DS in-order)
  _Float16* Ps = Qs;
#pragma unroll
  for (int mt = 0; mt < 4; ++mt)
#pragma unroll
    for (int nt = 0; nt < 4; ++nt)
#pragma unroll
      for (int r = 0; r < 8; ++r)
        Ps[(mt * 16 + r + hi * 8) * PLD + nt * 16 + n] = (_Float16)S[mt][nt][r];

  // Stage 3: O_h = P V -> OS column slice [wave*32, wave*32+32)
  v16h vf[2][2];
#pragma unroll
  for (int nt = 0; nt < 2; ++nt)
#pragma unroll
    for (int kt = 0; kt < 2; ++kt)
      vf[nt][kt] = ld_frag_rows(Vt + nt * 16 * VTLD + kt * 32, VTLD, lane);
  for (int mt = 0; mt < 4; ++mt) {
    v16h a0 = ld_frag_rows(Ps + mt * 16 * PLD, PLD, lane);
    v16h a1 = ld_frag_rows(Ps + mt * 16 * PLD + 32, PLD, lane);
#pragma unroll
    for (int nt = 0; nt < 2; ++nt) {
      v8f acc = {};
      acc = wmma16(a0, vf[nt][0], acc);
      acc = wmma16(a1, vf[nt][1], acc);
#pragma unroll
      for (int r = 0; r < 8; ++r)
        OS[(mt * 16 + r + hi * 8) * XLD + wave * HD + nt * 16 + n] =
            (_Float16)acc[r];
    }
  }
  __syncthreads();

  // Stage 4: out = O @ wproj + bproj ; each wave owns 32 output channels
  __builtin_prefetch(wproj_p + (size_t)(wave * 2 * 6) * 512 + lane * 64, 0, 1);
  v16h Bp[2][6];
#pragma unroll
  for (int nt = 0; nt < 2; ++nt)
#pragma unroll
    for (int kt = 0; kt < 6; ++kt)
      Bp[nt][kt] = ld_frag_packed(wproj_p, (wave * 2 + nt) * 6 + kt, lane);
  float* ob = out + (size_t)blk * NTOK * DIMC;
  const float bp0 = bproj[wave * HD + n], bp1 = bproj[wave * HD + 16 + n];
  for (int mt = 0; mt < 4; ++mt) {
    v16h a[6];
#pragma unroll
    for (int kt = 0; kt < 6; ++kt)
      a[kt] = ld_frag_rows(OS + mt * 16 * XLD + kt * 32, XLD, lane);
    v8f acc0 = {}, acc1 = {};
#pragma unroll
    for (int kt = 0; kt < 6; ++kt) {
      acc0 = wmma16(a[kt], Bp[0][kt], acc0);
      acc1 = wmma16(a[kt], Bp[1][kt], acc1);
    }
#pragma unroll
    for (int r = 0; r < 8; ++r) {
      const int row = mt * 16 + r + hi * 8;
      if (row < NTOK) {
        ob[row * DIMC + wave * HD + n]      = acc0[r] + bp0;
        ob[row * DIMC + wave * HD + 16 + n] = acc1[r] + bp1;
      }
    }
  }
}

extern "C" void kernel_launch(void* const* d_in, const int* in_sizes, int n_in,
                              void* d_out, int out_size, void* d_ws,
                              size_t ws_size, hipStream_t stream) {
  (void)in_sizes; (void)n_in; (void)out_size; (void)ws_size;
  const float* x     = (const float*)d_in[0];
  const float* wqkv  = (const float*)d_in[1];
  const float* bqkv  = (const float*)d_in[2];
  const float* wproj = (const float*)d_in[3];
  const float* bproj = (const float*)d_in[4];
  const float* btab  = (const float*)d_in[5];
  const int*   ridx  = (const int*)d_in[6];

  char* ws = (char*)d_ws;
  _Float16* wqkv_p  = (_Float16*)ws;                       // 110592 halfs, packed
  _Float16* wproj_p = (_Float16*)(ws + 221184);            // 36864 halfs, packed
  float*    biasf   = (float*)(ws + 221184 + 73728);       // 24576 floats

  winattn_prep<<<dim3(432), dim3(256), 0, stream>>>(wqkv, wproj, btab, ridx,
                                                    wqkv_p, wproj_p, biasf);

  const size_t shbytes =
      (size_t)(NP * XLD + NH * HCH + NP * XLD) * sizeof(_Float16);
  (void)hipFuncSetAttribute((const void*)winattn_kernel,
                            hipFuncAttributeMaxDynamicSharedMemorySize,
                            (int)shbytes);
  winattn_kernel<<<dim3(NWIN), dim3(192), shbytes, stream>>>(
      x, wqkv_p, bqkv, wproj_p, bproj, biasf, (float*)d_out);
}
